// Sparse_Conv_36876589203646
// MI455X (gfx1250) — matrix operand order repaired and checked by reading, not compiled, not run on hardware
//
#include <hip/hip_runtime.h>
#include <hip/hip_bf16.h>

// ---------------------------------------------------------------------------
// Masked 3x3 conv as implicit GEMM on CDNA5 WMMA (bf16 -> f32 accumulate).
//   M = C_out (256), N = B*H*W (204800), K = C_in*9 (2304)
// Pass 1: x_bf16[b][y][x][c] = bf16(ori[b][c][y][x] * mask[b][y][x])   (NHWC)
// Pass 2: w_bf16[co][tap][c] = bf16(weight[co][c][ky][kx])             (K-major)
// Pass 3: block = 8 waves = all 8 co-tiles of one 32-wide spatial tile.
//         Activation row-slab (34 cols x 256 ch, halo zero-filled) staged in
//         LDS via global_load_async_to_lds_b128 (ASYNCcnt-tracked),
//         double-buffered over dy. Per wave: 32x32 register tile, 4 WMMA/chunk.
// ---------------------------------------------------------------------------

typedef __attribute__((ext_vector_type(16))) __bf16 v16bf;
typedef __attribute__((ext_vector_type(8)))  __bf16 v8bf;
typedef __attribute__((ext_vector_type(8)))  float  v8f;
typedef __attribute__((ext_vector_type(4)))  int    v4i;

static constexpr int Hc   = 160;
static constexpr int Wc   = 160;
static constexpr int CIN  = 256;
static constexpr int COUT = 256;
static constexpr int NB   = 8;
static constexpr int XT16 = Wc / 16;          // 10 (prep pass x-tiles)
static constexpr int XT32 = Wc / 32;          // 5  (conv pass x-tiles)
static constexpr int SCOLS = 34;              // 32 + 2 halo columns
static constexpr int CP    = 264;             // padded channel stride (bank-safe)
static constexpr size_t XBF_BYTES = (size_t)NB * Hc * Wc * CIN * 2;  // 104857600

#if defined(__has_builtin)
#if __has_builtin(__builtin_amdgcn_global_load_async_to_lds_b128) && \
    __has_builtin(__builtin_amdgcn_s_wait_asynccnt)
#define HAVE_ASYNC_LDS 1
#endif
#endif
#ifndef HAVE_ASYNC_LDS
#define HAVE_ASYNC_LDS 0
#endif

#if HAVE_ASYNC_LDS
typedef __attribute__((address_space(1))) v4i as1_v4i;   // global int4
typedef __attribute__((address_space(3))) v4i as3_v4i;   // LDS int4
#endif

// Stage 16 bytes (8 bf16) from global to LDS.
__device__ __forceinline__ void stage16(const __bf16* g, __bf16* l) {
#if HAVE_ASYNC_LDS
    __builtin_amdgcn_global_load_async_to_lds_b128((as1_v4i*)g, (as3_v4i*)l, 0, 0);
#else
    *(v8bf*)l = *(const v8bf*)g;
#endif
}

__device__ __forceinline__ void stage_fence() {
#if HAVE_ASYNC_LDS
    __builtin_amdgcn_s_wait_asynccnt(0);
#endif
}

// ---------------------------------------------------------------------------
// Pass 1: mask-fuse + NCHW(f32) -> NHWC(bf16) via LDS tile transpose.
// ---------------------------------------------------------------------------
__global__ __launch_bounds__(256) void prep_x_kernel(const float* __restrict__ ori,
                                                     const int*   __restrict__ mask,
                                                     __bf16*      __restrict__ xout) {
    __shared__ __bf16 tile[CIN][17];

    int blk = blockIdx.x;                         // b*H*XT16 + y*XT16 + xt
    const int xt = blk % XT16;  blk /= XT16;
    const int y  = blk % Hc;
    const int b  = blk / Hc;
    const int x0 = xt * 16;

    const int t  = threadIdx.x;
    const int xs = t & 15;
    const int cs = t >> 4;

    const float m = (float)mask[((size_t)b * Hc + y) * Wc + x0 + xs];
    const float* obase = ori + (((size_t)b * CIN) * Hc + y) * Wc + x0 + xs;

    #pragma unroll
    for (int p = 0; p < 16; ++p) {
        const int c = p * 16 + cs;
        tile[c][xs] = (__bf16)(obase[(size_t)c * Hc * Wc] * m);
    }
    __syncthreads();

    __bf16* wr = xout + (((size_t)b * Hc + y) * Wc + x0) * CIN + t;
    #pragma unroll
    for (int x2 = 0; x2 < 16; ++x2)
        wr[(size_t)x2 * CIN] = tile[t][x2];
}

// ---------------------------------------------------------------------------
// Pass 2: OIHW f32 -> [co][tap][c] bf16 (K = tap*256 + c contiguous).
// ---------------------------------------------------------------------------
__global__ __launch_bounds__(256) void prep_w_kernel(const float* __restrict__ w,
                                                     __bf16*      __restrict__ w2) {
    const int i   = blockIdx.x * 256 + threadIdx.x;   // (co*9 + tap)*256 + c
    const int c   = i & 255;
    const int r   = i >> 8;
    const int tap = r % 9;
    const int co  = r / 9;
    w2[i] = (__bf16)w[((size_t)co * CIN + c) * 9 + tap];
}

// ---------------------------------------------------------------------------
// Pass 3: WMMA implicit-GEMM conv with LDS-staged activations.
// ---------------------------------------------------------------------------
__global__ __launch_bounds__(256) void conv_wmma_kernel(const __bf16* __restrict__ xin,
                                                        const __bf16* __restrict__ w2,
                                                        const float*  __restrict__ bias,
                                                        float*        __restrict__ out) {
    __shared__ __bf16 slab[2][SCOLS * CP];        // 2 x 17952 B, double-buffered

    const int t    = threadIdx.x;
    const int lane = t & 31;
    const int wave = t >> 5;                      // = co tile (0..7)

    int blk = blockIdx.x;                         // spatial tile: b*H*XT32 + ...
    const int xt = blk % XT32;  blk /= XT32;
    const int y  = blk % Hc;
    const int b  = blk / Hc;
    const int x0 = xt * 32;
    const int cb = wave * 32;

    const int half = lane >> 4;                   // lane half (K interleave)
    const int idx  = lane & 15;                   // A: row M;  B: column N
    const int kb   = half * 8;                    // A K base is 2*kb within chunk

    // Two A row streams: co = cb+idx and cb+16+idx (K-major weight rows)
    const __bf16* wrow0 = w2 + (size_t)(cb + idx)      * (9 * CIN);
    const __bf16* wrow1 = w2 + (size_t)(cb + 16 + idx) * (9 * CIN);

    v8bf z8;
    #pragma unroll
    for (int i = 0; i < 8; ++i) z8[i] = (__bf16)0.0f;

    v8f acc00 = {}, acc01 = {}, acc10 = {}, acc11 = {};

    for (int dyi = 0; dyi < 3; ++dyi) {
        const int yy = y + dyi - 1;
        __bf16* sl = slab[dyi & 1];
        const bool rowok = (unsigned)yy < (unsigned)Hc;
        const __bf16* xrow = xin + ((size_t)(b * Hc + yy) * Wc) * CIN;

        // ---- stage 34 cols x 256 ch (zero-fill halo/OOB); 1088 16B units ----
        for (int u = t; u < SCOLS * 32; u += 256) {
            const int col = u >> 5;               // 0..33
            const int ch  = (u & 31) << 3;        // 0..248
            const int xg  = x0 - 1 + col;
            __bf16* dst = sl + col * CP + ch;
            if (rowok && (unsigned)xg < (unsigned)Wc)
                stage16(xrow + (size_t)xg * CIN + ch, dst);
            else
                *(v8bf*)dst = z8;
        }
        stage_fence();
        __syncthreads();   // one barrier per dy; double-buffer covers the WAR

        // ---- 3 taps share this slab ----
        #pragma unroll
        for (int dx = 0; dx < 3; ++dx) {
            const int tap = dyi * 3 + dx;
            const __bf16* wtap0 = wrow0 + tap * CIN;
            const __bf16* wtap1 = wrow1 + tap * CIN;
            const __bf16* bcol0 = sl + (idx + dx)      * CP;   // B0: x = x0-1+idx+dx
            const __bf16* bcol1 = sl + (idx + 16 + dx) * CP;   // B1: x = +16

            #pragma unroll
            for (int c0 = 0; c0 < CIN; c0 += 32) {
                // A fragments (global, L2-resident): K = {2kb..2kb+15} (16 contiguous)
                union { v16bf v; v8bf h[2]; } A0, A1, B0, B1;
                A0.h[0] = *(const v8bf*)(wtap0 + c0 + kb * 2);
                A0.h[1] = *(const v8bf*)(wtap0 + c0 + kb * 2 + 8);
                A1.h[0] = *(const v8bf*)(wtap1 + c0 + kb * 2);
                A1.h[1] = *(const v8bf*)(wtap1 + c0 + kb * 2 + 8);

                // B fragments (LDS): 16 contiguous K, lane-half selects K block
                B0.h[0] = *(const v8bf*)(bcol0 + c0 + half * 16);
                B0.h[1] = *(const v8bf*)(bcol0 + c0 + half * 16 + 8);
                B1.h[0] = *(const v8bf*)(bcol1 + c0 + half * 16);
                B1.h[1] = *(const v8bf*)(bcol1 + c0 + half * 16 + 8);

                acc00 = __builtin_amdgcn_wmma_f32_16x16x32_bf16(false, A0.v, false, B0.v,
                            (short)0, acc00, false, false);
                acc01 = __builtin_amdgcn_wmma_f32_16x16x32_bf16(false, A0.v, false, B1.v,
                            (short)0, acc01, false, false);
                acc10 = __builtin_amdgcn_wmma_f32_16x16x32_bf16(false, A1.v, false, B0.v,
                            (short)0, acc10, false, false);
                acc11 = __builtin_amdgcn_wmma_f32_16x16x32_bf16(false, A1.v, false, B1.v,
                            (short)0, acc11, false, false);
            }
        }
    }

    // Epilogue. D layout: column N = idx, VGPR r -> row (+half*8 + r).
    const v8f bv0 = *(const v8f*)(bias + cb + half * 8);
    const v8f bv1 = *(const v8f*)(bias + cb + 16 + half * 8);
    float* ob00 = out + (((size_t)b * COUT + cb + half * 8)      * Hc + y) * Wc + x0 + idx;
    float* ob10 = out + (((size_t)b * COUT + cb + 16 + half * 8) * Hc + y) * Wc + x0 + idx;
    #pragma unroll
    for (int r = 0; r < 8; ++r) {
        const size_t ro = (size_t)r * Hc * Wc;
        ob00[ro]      = acc00[r] + bv0[r];   // co tile 0, x tile 0
        ob00[ro + 16] = acc01[r] + bv0[r];   // co tile 0, x tile 1
        ob10[ro]      = acc10[r] + bv1[r];   // co tile 1, x tile 0
        ob10[ro + 16] = acc11[r] + bv1[r];   // co tile 1, x tile 1
    }
}

// ---------------------------------------------------------------------------
extern "C" void kernel_launch(void* const* d_in, const int* in_sizes, int n_in,
                              void* d_out, int out_size, void* d_ws, size_t ws_size,
                              hipStream_t stream) {
    const int*   mask = (const int*)  d_in[0];
    const float* ori  = (const float*)d_in[1];
    const float* wgt  = (const float*)d_in[2];
    const float* bias = (const float*)d_in[3];
    float* out = (float*)d_out;

    char* ws = (char*)d_ws;
    __bf16* xbf = (__bf16*)ws;                    // 104,857,600 B  (NHWC bf16 x)
    __bf16* wbf = (__bf16*)(ws + XBF_BYTES);      //   1,179,648 B  (repacked bf16 w)

    // Pass 1: 8*160*10 = 12800 tile-transpose blocks
    prep_x_kernel<<<NB * Hc * XT16, 256, 0, stream>>>(ori, mask, xbf);
    // Pass 2: 589824 weight elements
    prep_w_kernel<<<(COUT * 9 * CIN) / 256, 256, 0, stream>>>(wgt, wbf);
    // Pass 3: 6400 spatial-tile blocks, 8 waves = 8 co tiles each
    conv_wmma_kernel<<<NB * Hc * XT32, 256, 0, stream>>>(xbf, wbf, bias, out);
}
